// TopKPool_18227841204839
// MI455X (gfx1250) — matrix operand order repaired and checked by reading, not compiled, not run on hardware
//
#include <hip/hip_runtime.h>
#include <hip/hip_bf16.h>

typedef __attribute__((ext_vector_type(16))) _Float16 v16h;
typedef __attribute__((ext_vector_type(8)))  float    v8f;

#define HIDDEN 256
#define H2     128
#define NPG    1024
#define KSEL   512

union FragU { unsigned int u[8]; v16h h; };
union H16U  { _Float16 f; unsigned short u; };

// ---------------------------------------------------------------------------
// Kernel 0 (one-shot prepass): convert W1 [256,128] f32 -> f16 and store in
// WMMA B-fragment layout in global memory (64KB, stays L2/WGP$-resident).
// Layout: uint index = t*2048 + nt*256 + lane*8 + j, where
//   t = k>>5 (K-tile), kin = k&31, lanes 0-15 hold K 0..15 (uint j = pair
//   2j,2j+1), lanes 16-31 hold K 16..31; lane = (kin>=16?16:0) + (n&15),
//   nt = n>>4.
// ---------------------------------------------------------------------------
__global__ void __launch_bounds__(256)
swizzle_w1_kernel(const float* __restrict__ W1, unsigned int* __restrict__ w1sw)
{
    int e = blockIdx.x * 256 + threadIdx.x;   // 0..32767
    int k = e >> 7;                           // 0..255
    int n = e & 127;                          // 0..127
    H16U cv; cv.f = (_Float16)W1[k * H2 + n];
    int t   = k >> 5;
    int kin = k & 31;
    int hi  = kin >> 4;
    int j   = (kin & 15) >> 1;
    int hlf = kin & 1;
    int l   = hi * 16 + (n & 15);
    int nt  = n >> 4;
    ((unsigned short*)w1sw)[(t * 2048 + nt * 256 + l * 8 + j) * 2 + hlf] = cv.u;
}

// ---------------------------------------------------------------------------
// Kernel 1: LayerNorm + scorer MLP via WMMA. 128 threads = 4 waves,
// 64 nodes per block (16 nodes per wave -> one 16-row WMMA A tile).
// B fragments come straight from the pre-swizzled global W1 (cache-hit
// global_load_b128), so the K-loop has no barriers at all.
// ---------------------------------------------------------------------------
__global__ void __launch_bounds__(128)
topk_score_kernel(const float* __restrict__ x,
                  const float* __restrict__ gamma,
                  const float* __restrict__ beta,
                  const unsigned int* __restrict__ w1sw, // pre-swizzled f16 W1
                  const float* __restrict__ b1,          // [128]
                  const float* __restrict__ W2,          // [128]
                  const float* __restrict__ b2,          // [1]
                  float* __restrict__ scores)            // [total_nodes]
{
    // Per-wave-private xn staging in A-fragment layout (2x ds_load_b128/frag)
    __shared__ __align__(16) unsigned int sA[4 * 2048];  // 32KB

    const int tid   = threadIdx.x;
    const int wave  = tid >> 5;
    const int lane  = tid & 31;
    const int node0 = blockIdx.x * 64 + wave * 16;

    // ---- LayerNorm: 16 rows per wave, each lane owns 8 columns ----
    float g8[8], bt8[8];
    #pragma unroll
    for (int c = 0; c < 8; ++c) {
        g8[c]  = gamma[lane * 8 + c];
        bt8[c] = beta[lane * 8 + c];
    }

    unsigned short* sAh = (unsigned short*)&sA[wave * 2048];

    for (int row = 0; row < 16; ++row) {
        const float* xr = x + (size_t)(node0 + row) * HIDDEN + lane * 8;
        float4 v0 = ((const float4*)xr)[0];
        float4 v1 = ((const float4*)xr)[1];
        float e[8] = {v0.x, v0.y, v0.z, v0.w, v1.x, v1.y, v1.z, v1.w};
        float s = 0.f, sq = 0.f;
        #pragma unroll
        for (int c = 0; c < 8; ++c) { s += e[c]; sq += e[c] * e[c]; }
        #pragma unroll
        for (int m = 16; m >= 1; m >>= 1) {   // wave32 tree reduction
            s  += __shfl_xor(s,  m);
            sq += __shfl_xor(sq, m);
        }
        float mu   = s * (1.f / 256.f);
        float var  = sq * (1.f / 256.f) - mu * mu;
        float rstd = rsqrtf(var + 1e-5f);
        #pragma unroll
        for (int c = 0; c < 8; ++c) {
            float xn = (e[c] - mu) * rstd * g8[c] + bt8[c];
            H16U cv; cv.f = (_Float16)xn;
            // A-fragment swizzle (16-bit A 16x32 tile, contiguous k order, same as W1 B layout):
            // lanes 0-15 hold K 0..15; lanes 16-31 hold K 16..31
            int k   = lane * 8 + c;
            int t   = k >> 5;          // K-tile
            int kin = k & 31;          // K within tile
            int hi  = kin >> 4;
            int j   = (kin & 15) >> 1;
            int hlf = kin & 1;
            int l   = hi * 16 + row;
            sAh[(t * 256 + l * 8 + j) * 2 + hlf] = cv.u;
        }
    }
    __syncthreads();   // fence LDS stores vs fragment loads (aliasing safety)

    v8f acc[8];
    #pragma unroll
    for (int nt = 0; nt < 8; ++nt) acc[nt] = (v8f){0.f,0.f,0.f,0.f,0.f,0.f,0.f,0.f};

    // ---- GEMM: h[16x128] = xn[16x256] @ W1[256x128], K in 8 steps of 32 ----
    for (int t = 0; t < 8; ++t) {
        FragU a;
        const uint4* pa = (const uint4*)&sA[wave * 2048 + t * 256 + lane * 8];
        ((uint4*)a.u)[0] = pa[0];
        ((uint4*)a.u)[1] = pa[1];

        const uint4* pw = ((const uint4*)w1sw) + (t * 512 + lane * 2);
        #pragma unroll
        for (int nt = 0; nt < 8; ++nt) {
            FragU bf;
            ((uint4*)bf.u)[0] = pw[nt * 64];      // nt*256 uints
            ((uint4*)bf.u)[1] = pw[nt * 64 + 1];
            acc[nt] = __builtin_amdgcn_wmma_f32_16x16x32_f16(
                false, a.h, false, bf.h, (short)0, acc[nt], false, false);
        }
    }

    // ---- Epilogue: +b1, SiLU, dot with W2, +b2 -> one score per node ----
    // C/D layout: tile nt, VGPR v: lanes 0-15 -> (M=v, N=nt*16+lane);
    //                              lanes 16-31 -> (M=v+8, N=nt*16+lane-16)
    float p[8];
    #pragma unroll
    for (int v = 0; v < 8; ++v) p[v] = 0.f;
    const int ncol = lane & 15;
    #pragma unroll
    for (int nt = 0; nt < 8; ++nt) {
        int n = nt * 16 + ncol;
        float b1v = b1[n];
        float w2v = W2[n];
        #pragma unroll
        for (int v = 0; v < 8; ++v) {
            float hv = acc[nt][v] + b1v;
            float sl = hv / (1.f + __expf(-hv));   // SiLU
            p[v] += sl * w2v;
        }
    }
    #pragma unroll
    for (int m = 8; m >= 1; m >>= 1) {   // reduce within each 16-lane half
        #pragma unroll
        for (int v = 0; v < 8; ++v) p[v] += __shfl_xor(p[v], m);
    }
    float b2v = b2[0];
    if (lane == 0) {
        #pragma unroll
        for (int v = 0; v < 8; ++v) scores[node0 + v] = p[v] + b2v;
    }
    if (lane == 16) {
        #pragma unroll
        for (int v = 0; v < 8; ++v) scores[node0 + 8 + v] = p[v] + b2v;
    }
}

// ---------------------------------------------------------------------------
// Kernel 2: per-graph top-512 of 1024 via LDS bitonic sort (desc, tie -> lower
// index, matching jax.lax.top_k), then fused row gather. One block per graph.
// ---------------------------------------------------------------------------
__global__ void __launch_bounds__(1024)
topk_select_gather_kernel(const float* __restrict__ x,
                          const float* __restrict__ scores,
                          float* __restrict__ out_pool,  // [B*512, 256]
                          float* __restrict__ out_idx,   // [B*512] (as float)
                          float* __restrict__ out_k)     // [1]
{
    __shared__ float sval[NPG];
    __shared__ int   sidx[NPG];
    const int tid = threadIdx.x;
    const int g   = blockIdx.x;

    sval[tid] = scores[g * NPG + tid];
    sidx[tid] = tid;
    __syncthreads();

    for (int ksz = 2; ksz <= NPG; ksz <<= 1) {
        for (int j = ksz >> 1; j > 0; j >>= 1) {
            int partner = tid ^ j;
            if (partner > tid) {
                float va = sval[tid],  vb = sval[partner];
                int   ia = sidx[tid],  ib = sidx[partner];
                bool aFirst = (va > vb) || (va == vb && ia < ib);
                bool desc   = ((tid & ksz) == 0);
                if (desc ? !aFirst : aFirst) {
                    sval[tid] = vb; sval[partner] = va;
                    sidx[tid] = ib; sidx[partner] = ia;
                }
            }
            __syncthreads();
        }
    }

    if (tid < KSEL) {
        int gi = g * NPG + sidx[tid];
        out_idx[g * KSEL + tid] = (float)gi;
    }
    __syncthreads();

    // Gather 512 rows x 256 floats = 32768 float4, 32 per thread
    #pragma unroll 4
    for (int i = 0; i < 32; ++i) {
        int flat = tid + i * 1024;
        int r    = flat >> 6;     // selected row 0..511
        int c4   = flat & 63;     // float4 column
        int gi   = g * NPG + sidx[r];
        float4 v = ((const float4*)(x + (size_t)gi * HIDDEN))[c4];
        ((float4*)(out_pool + (size_t)(g * KSEL + r) * HIDDEN))[c4] = v;
    }

    if (g == 0 && tid == 0) out_k[0] = (float)KSEL;
}

extern "C" void kernel_launch(void* const* d_in, const int* in_sizes, int n_in,
                              void* d_out, int out_size, void* d_ws, size_t ws_size,
                              hipStream_t stream) {
    const float* x     = (const float*)d_in[0];
    const float* gamma = (const float*)d_in[1];
    const float* beta  = (const float*)d_in[2];
    const float* W1    = (const float*)d_in[3];
    const float* b1    = (const float*)d_in[4];
    const float* W2    = (const float*)d_in[5];
    const float* b2    = (const float*)d_in[6];
    // d_in[7] = nodes_per_graph (device scalar); fixed at 1024 per reference.

    const int total_nodes = in_sizes[0] / HIDDEN;   // 262144
    const int batch       = total_nodes / NPG;      // 256

    unsigned int* w1sw = (unsigned int*)d_ws;       // 16384 uints = 64KB
    float* scores      = (float*)d_ws + 16384;      // total_nodes floats (1 MB)
    float* out         = (float*)d_out;
    size_t poolElems   = (size_t)batch * KSEL * HIDDEN;
    float* out_idx     = out + poolElems;
    float* out_k       = out_idx + (size_t)batch * KSEL;

    swizzle_w1_kernel<<<128, 256, 0, stream>>>(W1, w1sw);
    topk_score_kernel<<<total_nodes / 64, 128, 0, stream>>>(
        x, gamma, beta, w1sw, b1, W2, b2, scores);
    topk_select_gather_kernel<<<batch, 1024, 0, stream>>>(
        x, scores, out, out_idx, out_k);
}
